// CausalSelfAttention_24455543783709
// MI455X (gfx1250) — compile-verified, matrix operand order checked by reading, not run on hardware
//
#include <hip/hip_runtime.h>
#include <hip/hip_bf16.h>

// ---------------------------------------------------------------------------
// CausalSelfAttention forward for MI455X (gfx1250, wave32, WMMA bf16 path)
// B=8, T=1024, C=768, H=12, HD=64
// ---------------------------------------------------------------------------

#define Bn 8
#define Tn 1024
#define Cn 768
#define Hn 12
#define HDn 64

typedef __attribute__((ext_vector_type(16))) __bf16 v16bf;
typedef __attribute__((ext_vector_type(2)))  __bf16 v2bf;
typedef __attribute__((ext_vector_type(8)))  float  v8f;
typedef __attribute__((ext_vector_type(2)))  float  v2f;
typedef int v4i __attribute__((vector_size(16)));

union BFrag {
    unsigned u[8];
    v16bf    v;
};

// native bf16 converts (v_cvt_pk_bf16_f32 on gfx1250)
__device__ __forceinline__ unsigned short f2bf(float f) {
    __bf16 b = (__bf16)f;
    return __builtin_bit_cast(unsigned short, b);
}
__device__ __forceinline__ unsigned pack2(float lo, float hi) {
    v2f f; f.x = lo; f.y = hi;
    v2bf b = __builtin_convertvector(f, v2bf);
    return __builtin_bit_cast(unsigned, b);
}

// ---- async global->LDS copy (gfx1250 ASYNCcnt path) with safe fallback ----
#if __has_builtin(__builtin_amdgcn_global_load_async_to_lds_b128) && \
    __has_builtin(__builtin_amdgcn_s_wait_asynccnt)
#define ASYNC_LDS 1
#else
#define ASYNC_LDS 0
#endif

__device__ __forceinline__ void copy16B_g2l(unsigned short* ldst,
                                            const unsigned short* gsrc) {
#if ASYNC_LDS
    unsigned short* gs = const_cast<unsigned short*>(gsrc);
    __builtin_amdgcn_global_load_async_to_lds_b128(
        (__attribute__((address_space(1))) v4i*)gs,
        (__attribute__((address_space(3))) v4i*)ldst, 0, 0);
#else
    *(uint4*)ldst = *(const uint4*)gsrc;
#endif
}
__device__ __forceinline__ void async_wait() {
#if ASYNC_LDS
    __builtin_amdgcn_s_wait_asynccnt(0);
#endif
}

// K index for 16-bit A/B fragment VGPR v, wave half (lane>=16):
//   VGPR0..3: K = half*8 + 2*v ; VGPR4..7: +16   (pairs K, K+1 per dword)
__device__ __forceinline__ int frag_k(int v, int half) {
    return ((v >> 2) << 4) | (half << 3) | ((v & 3) << 1);
}

__device__ __forceinline__ v8f wmma_bf16(const BFrag& a, const BFrag& b, v8f c) {
    return __builtin_amdgcn_wmma_f32_16x16x32_bf16(
        /*neg_a=*/false, a.v, /*neg_b=*/false, b.v,
        /*c_mod=*/(short)0, c, /*reuse_a=*/false, /*reuse_b=*/false);
}

__device__ __forceinline__ v8f v8f_zero() {
    v8f z = {0.f, 0.f, 0.f, 0.f, 0.f, 0.f, 0.f, 0.f};
    return z;
}

// ---- DPP butterfly reduction within lane-groups of 16 (VALU only) ----
template <int CTRL>
__device__ __forceinline__ float fdpp(float x) {
    int r = __builtin_amdgcn_update_dpp(0, __float_as_int(x), CTRL,
                                        0xF, 0xF, true);
    return __int_as_float(r);
}
__device__ __forceinline__ float row16_max(float x) {
    x = fmaxf(x, fdpp<0xB1>(x));   // quad_perm {1,0,3,2}  (xor 1)
    x = fmaxf(x, fdpp<0x4E>(x));   // quad_perm {2,3,0,1}  (xor 2)
    x = fmaxf(x, fdpp<0x141>(x));  // row_half_mirror      (octet)
    x = fmaxf(x, fdpp<0x140>(x));  // row_mirror           (group of 16)
    return x;
}
__device__ __forceinline__ float row16_sum(float x) {
    x += fdpp<0xB1>(x);
    x += fdpp<0x4E>(x);
    x += fdpp<0x141>(x);
    x += fdpp<0x140>(x);
    return x;
}

// ---------------------------------------------------------------------------
// One-time conversions: f32 -> bf16 (elementwise), and transposed weights
// ---------------------------------------------------------------------------
__global__ __launch_bounds__(256) void cvt_bf16_kernel(
    const float* __restrict__ in, unsigned short* __restrict__ out, int n)
{
    int i = (blockIdx.x * 256 + threadIdx.x) * 2;   // n is even
    if (i < n) {
        *(unsigned*)(out + i) = pack2(in[i], in[i + 1]);
    }
}

// WT[n][k] = bf16(W[k][n]);  W is [K][N] row-major
__global__ __launch_bounds__(256) void transpose_cvt_kernel(
    const float* __restrict__ W, unsigned short* __restrict__ WT, int K, int N)
{
    int k = blockIdx.x * 256 + threadIdx.x;
    int n = blockIdx.y;
    WT[(size_t)n * K + k] = f2bf(W[(size_t)k * N + n]);
}

// ---------------------------------------------------------------------------
// GEMM: out[M,N] = A[M,K] * Bt[N,K]^T + bias[N]   (A, Bt bf16; out f32 or bf16)
// block = 256 threads (8 waves), tile 128x64, K-step 32, async LDS staging
// ---------------------------------------------------------------------------
template <bool OBF16>
__global__ __launch_bounds__(256) void gemm_bias_kernel(
    const unsigned short* __restrict__ A, const unsigned short* __restrict__ Bt,
    const float* __restrict__ bias, void* __restrict__ outp,
    int M, int N, int K)
{
    __shared__ __align__(16) unsigned short As[128 * 32];  // [m][k] bf16
    __shared__ __align__(16) unsigned short Bs[64 * 32];   // [n][k] bf16
    unsigned* Asd = (unsigned*)As;
    unsigned* Bsd = (unsigned*)Bs;

    const int tid  = threadIdx.x;
    const int lane = tid & 31;
    const int wave = tid >> 5;     // 0..7 -> 16-row strip
    const int half = lane >> 4;
    const int n16  = lane & 15;

    const int m0b = blockIdx.y * 128;
    const int n0b = blockIdx.x * 64;

    v8f acc[4];
#pragma unroll
    for (int f = 0; f < 4; ++f) acc[f] = v8f_zero();

    const int rA   = tid >> 1;          // 0..127 : A row this thread stages
    const int segA = (tid & 1) * 16;    // A k-segment (16 elements)
    const int nB   = tid & 63;          // B row (=output col) this thread stages
    const int segB = (tid >> 6) * 8;    // B k-segment (8 elements)

    for (int k0 = 0; k0 < K; k0 += 32) {
        // ---- stage tiles: contiguous bf16 copies (async to LDS) ----
        {
            const unsigned short* asrc = A + (size_t)(m0b + rA) * K + k0 + segA;
            copy16B_g2l(&As[rA * 32 + segA],     asrc);
            copy16B_g2l(&As[rA * 32 + segA + 8], asrc + 8);
            const unsigned short* bsrc = Bt + (size_t)(n0b + nB) * K + k0 + segB;
            copy16B_g2l(&Bs[nB * 32 + segB],     bsrc);
        }
        async_wait();
        __syncthreads();

        // ---- WMMA: this wave's 16-row strip x 64 cols ----
        BFrag af;
#pragma unroll
        for (int v = 0; v < 8; ++v)
            af.u[v] = Asd[(wave * 16 + n16) * 16 + (frag_k(v, half) >> 1)];
#pragma unroll
        for (int f = 0; f < 4; ++f) {
            BFrag bfx;
#pragma unroll
            for (int v = 0; v < 8; ++v)
                bfx.u[v] = Bsd[(f * 16 + n16) * 16 + (frag_k(v, half) >> 1)];
            acc[f] = wmma_bf16(af, bfx, acc[f]);
        }
        __syncthreads();
    }

    // ---- epilogue: D layout lane=n%16, vgpr v -> m = v + 8*(lane/16) ----
#pragma unroll
    for (int f = 0; f < 4; ++f) {
#pragma unroll
        for (int v = 0; v < 8; ++v) {
            int m = m0b + wave * 16 + v + 8 * half;
            int n = n0b + f * 16 + n16;
            float val = acc[f][v] + bias[n];
            if (OBF16)
                ((unsigned short*)outp)[(size_t)m * N + n] = f2bf(val);
            else
                ((float*)outp)[(size_t)m * N + n] = val;
        }
    }
}

// ---------------------------------------------------------------------------
// Flash attention: 128 threads (4 waves) per (b, h, 64-query tile).
// qkv is bf16 [B, T, 3C]; per token/head: q @ h*3HD, k @ +HD, v @ +2HD.
// Writes y as bf16 [B, T, C] for the projection GEMM.
// ---------------------------------------------------------------------------
__global__ __launch_bounds__(128) void attn_kernel(
    const unsigned short* __restrict__ qkv, unsigned short* __restrict__ ybf)
{
    __shared__ __align__(16) unsigned short Kt[32 * 64];      // [key][d] bf16
    __shared__ __align__(16) unsigned short Vt[64 * 32];      // [d][key] bf16
    __shared__ __align__(16) unsigned short Pt[4 * 16 * 32];  // per-wave [m][key]
    unsigned* Ktd = (unsigned*)Kt;
    unsigned* Vtd = (unsigned*)Vt;
    unsigned* Ptd = (unsigned*)Pt;

    const int tid  = threadIdx.x;
    const int lane = tid & 31;
    const int wave = tid >> 5;     // 0..3
    const int half = lane >> 4;
    const int n16  = lane & 15;

    const int qblk  = blockIdx.x * 64;
    const int qbase = qblk + wave * 16;       // this wave's query tile
    const int h     = blockIdx.y;
    const int b     = blockIdx.z;
    const int hd0   = h * 3 * HDn;
    const int pofs  = wave * 256;             // dword offset of wave's P slice

    // ---- Q fragments: bf16 loads, scale by 1/sqrt(HD)=0.125, repack ----
    BFrag aq[2];
    {
        const unsigned short* qptr =
            qkv + (size_t)(b * Tn + qbase + n16) * (3 * Cn) + hd0;
#pragma unroll
        for (int c = 0; c < 2; ++c) {
#pragma unroll
            for (int v = 0; v < 8; ++v) {
                int d = c * 32 + frag_k(v, half);
                unsigned raw = *(const unsigned*)(qptr + d);
                float lo = __uint_as_float(raw << 16) * 0.125f;
                float hi = __uint_as_float(raw & 0xFFFF0000u) * 0.125f;
                aq[c].u[v] = pack2(lo, hi);
            }
        }
    }

    v8f o[4];
#pragma unroll
    for (int f = 0; f < 4; ++f) o[f] = v8f_zero();
    float mrow[8], lrow[8];
#pragma unroll
    for (int v = 0; v < 8; ++v) { mrow[v] = -1e30f; lrow[v] = 0.f; }

    // staging assignment: thread -> (key, 16-wide d segment)
    const int skey = tid >> 2;
    const int sd0  = (tid & 3) * 16;

    const int kmax = qblk + 48 + 15;          // block-uniform loop bound
    for (int kbase = 0; kbase <= kmax; kbase += 32) {
        // ---- stage K (async contiguous) and V (transposed scatter) ----
        {
            const unsigned short* base =
                qkv + (size_t)(b * Tn + kbase + skey) * (3 * Cn) + hd0;
            const unsigned short* ksrc = base + HDn + sd0;
            copy16B_g2l(&Kt[skey * 64 + sd0],     ksrc);
            copy16B_g2l(&Kt[skey * 64 + sd0 + 8], ksrc + 8);
            const unsigned short* vsrc = base + 2 * HDn;
#pragma unroll
            for (int d = sd0; d < sd0 + 16; d += 2) {
                unsigned dv = *(const unsigned*)(vsrc + d);
                Vt[d * 32 + skey]       = (unsigned short)(dv & 0xFFFFu);
                Vt[(d + 1) * 32 + skey] = (unsigned short)(dv >> 16);
            }
        }
        async_wait();
        __syncthreads();

        if (kbase <= qbase + 15) {   // wave-uniform; EXEC stays all-ones
            // ---- S = Q * K^T (two 16x16 column tiles, K-dim 64) ----
            v8f s[2];
#pragma unroll
            for (int t = 0; t < 2; ++t) {
                const int n0 = t * 16;
                v8f acc = v8f_zero();
#pragma unroll
                for (int c = 0; c < 2; ++c) {
                    BFrag bk;
#pragma unroll
                    for (int v = 0; v < 8; ++v)
                        bk.u[v] = Ktd[(n0 + n16) * 32 +
                                      ((c * 32 + frag_k(v, half)) >> 1)];
                    acc = wmma_bf16(aq[c], bk, acc);
                }
                s[t] = acc;
            }

            // ---- causal mask ----
#pragma unroll
            for (int t = 0; t < 2; ++t) {
#pragma unroll
                for (int v = 0; v < 8; ++v) {
                    int mg = qbase + v + 8 * half;
                    int kg = kbase + t * 16 + n16;
                    if (kg > mg) s[t][v] = -1e30f;
                }
            }

            // ---- online softmax (DPP reductions over lane groups of 16) ----
            float alpha[8];
#pragma unroll
            for (int v = 0; v < 8; ++v) {
                float tm   = row16_max(fmaxf(s[0][v], s[1][v]));
                float mnew = fmaxf(mrow[v], tm);
                alpha[v]   = __expf(mrow[v] - mnew);
                mrow[v]    = mnew;
                float p0 = __expf(s[0][v] - mnew); s[0][v] = p0;
                float p1 = __expf(s[1][v] - mnew); s[1][v] = p1;
                float r  = row16_sum(p0 + p1);
                lrow[v]  = lrow[v] * alpha[v] + r;
            }
#pragma unroll
            for (int f = 0; f < 4; ++f)
#pragma unroll
                for (int v = 0; v < 8; ++v) o[f][v] *= alpha[v];

            // ---- re-layout P (wave-private LDS slice, no barrier needed) ----
#pragma unroll
            for (int v = 0; v < 8; ++v) {
                int mv = v + 8 * half;
                Pt[pofs * 2 + mv * 32 + n16]      = f2bf(s[0][v]);
                Pt[pofs * 2 + mv * 32 + 16 + n16] = f2bf(s[1][v]);
            }

            BFrag ap;
#pragma unroll
            for (int v = 0; v < 8; ++v)
                ap.u[v] = Ptd[pofs + n16 * 16 + (frag_k(v, half) >> 1)];

            // ---- O += P * V (4 column groups of head dim) ----
#pragma unroll
            for (int f = 0; f < 4; ++f) {
                BFrag bv;
#pragma unroll
                for (int v = 0; v < 8; ++v)
                    bv.u[v] = Vtd[(f * 16 + n16) * 16 + (frag_k(v, half) >> 1)];
                o[f] = wmma_bf16(ap, bv, o[f]);
            }
        }
        __syncthreads();   // all waves done with Kt/Vt before restaging
    }

    // ---- finalize: O /= l, store bf16 into y[b, q, h*HD + d] ----
    float inv[8];
#pragma unroll
    for (int v = 0; v < 8; ++v)
        inv[v] = lrow[v] > 0.f ? 1.0f / lrow[v] : 0.f;
#pragma unroll
    for (int f = 0; f < 4; ++f) {
#pragma unroll
        for (int v = 0; v < 8; ++v) {
            int mv = v + 8 * half;
            size_t idx = (size_t)(b * Tn + qbase + mv) * Cn +
                         h * HDn + f * 16 + n16;
            ybf[idx] = f2bf(o[f][v] * inv[v]);
        }
    }
}

// ---------------------------------------------------------------------------
extern "C" void kernel_launch(void* const* d_in, const int* in_sizes, int n_in,
                              void* d_out, int out_size, void* d_ws, size_t ws_size,
                              hipStream_t stream) {
    (void)in_sizes; (void)n_in; (void)out_size; (void)ws_size;

    const float* x      = (const float*)d_in[0];  // [B,T,C]
    const float* W_attn = (const float*)d_in[1];  // [C,3C]
    const float* b_attn = (const float*)d_in[2];  // [3C]
    const float* W_proj = (const float*)d_in[3];  // [C,C]
    const float* b_proj = (const float*)d_in[4];  // [C]
    float*       out    = (float*)d_out;          // [B,T,C]

    const int M = Bn * Tn;      // 8192

    // workspace layout (all bf16)
    unsigned short* xbf   = (unsigned short*)d_ws;           // [M, C]
    unsigned short* wat   = xbf + (size_t)M * Cn;            // [3C, C] (W_attn^T)
    unsigned short* wpt   = wat + (size_t)3 * Cn * Cn;       // [C, C]  (W_proj^T)
    unsigned short* qkvbf = wpt + (size_t)Cn * Cn;           // [M, 3C]
    unsigned short* ybf   = qkvbf + (size_t)M * 3 * Cn;      // [M, C]

    // 0) one-time conversions
    cvt_bf16_kernel<<<(M * Cn / 2 + 255) / 256, 256, 0, stream>>>(
        x, xbf, M * Cn);
    transpose_cvt_kernel<<<dim3(Cn / 256, 3 * Cn), 256, 0, stream>>>(
        W_attn, wat, Cn, 3 * Cn);
    transpose_cvt_kernel<<<dim3(Cn / 256, Cn), 256, 0, stream>>>(
        W_proj, wpt, Cn, Cn);

    // 1) qkv = x @ W_attn + b_attn   (bf16 out)
    gemm_bias_kernel<true><<<dim3((3 * Cn) / 64, M / 128), 256, 0, stream>>>(
        xbf, wat, b_attn, (void*)qkvbf, M, 3 * Cn, Cn);

    // 2) flash attention per (b, h, 64-query block)
    attn_kernel<<<dim3(Tn / 64, Hn, Bn), 128, 0, stream>>>(qkvbf, ybf);

    // 3) out = y @ W_proj + b_proj   (f32 out)
    gemm_bias_kernel<false><<<dim3(Cn / 64, M / 128), 256, 0, stream>>>(
        ybf, wpt, b_proj, (void*)out, M, Cn, Cn);
}
